// GemmaAttention_66159676228240
// MI455X (gfx1250) — compile-verified, matrix operand order checked by reading, not run on hardware
//
#include <hip/hip_runtime.h>
#include <hip/hip_bf16.h>
#include <math.h>

// ---- problem constants (match reference) ----
constexpr int Bc   = 4;
constexpr int Sc   = 2048;
constexpr int HIDc = 2048;
constexpr int NHc  = 8;
constexpr int NKVc = 1;
constexpr int HDc  = 256;

typedef __bf16 bf16_t;
typedef bf16_t v16bf __attribute__((ext_vector_type(16)));
typedef float  v8f   __attribute__((ext_vector_type(8)));

__device__ __forceinline__ unsigned short f2bf(float x) {
  unsigned u = __builtin_bit_cast(unsigned, x);
  u = (u + 0x7FFFu + ((u >> 16) & 1u)) >> 16;   // round-to-nearest-even
  return (unsigned short)u;
}
__device__ __forceinline__ float bf2f(unsigned short h) {
  unsigned u = ((unsigned)h) << 16;
  return __builtin_bit_cast(float, u);
}

// ---- CDNA5 async global->LDS copy (ASYNCcnt path, ISA 10.7/15.18) ----
// Generic shared pointers carry the LDS offset in their low 32 bits
// (aperture form {SHARED_BASE, offset}), so truncation yields the DS address.
__device__ __forceinline__ void async_copy_b128(const void* lds_dst, const void* gsrc) {
  unsigned lds_off = (unsigned)(uintptr_t)lds_dst;
  unsigned long long gaddr = (unsigned long long)(uintptr_t)gsrc;
  asm volatile("global_load_async_to_lds_b128 %0, %1, off"
               :: "v"(lds_off), "v"(gaddr) : "memory");
}
__device__ __forceinline__ void wait_async0() {
  asm volatile("s_wait_asynccnt 0" ::: "memory");
}

union FragU { v16bf v; unsigned u[8]; };

// 16x32 bf16 fragment in the CDNA5 WMMA A/B register layout from a row-major
// (16 rows x >=32 cols) region. lane L: row = L%16, half = L/16; VGPR v holds
// packed K pair at k = 2v + (v>=4 ? 8 : 0) + (half ? 8 : 0)  (ISA 7.12.2).
// kofs runs 0..6 then 16..22 (+half*8): the 8 b32 reads fold into 2 b128.
__device__ __forceinline__ v16bf load_frag(const unsigned short* base, int stride, int lane) {
  FragU f;
  const int r  = lane & 15;
  const int hi = (lane >> 4) & 1;
  const unsigned short* rp = base + (size_t)r * stride;
#pragma unroll
  for (int vv = 0; vv < 8; ++vv) {
    const int kofs = 2 * vv + ((vv >= 4) ? 8 : 0) + (hi ? 8 : 0);
    f.u[vv] = *(const unsigned*)(rp + kofs);
  }
  return f.v;
}

__device__ __forceinline__ v8f wmma_bf16(v16bf a, v16bf b, v8f c) {
  return __builtin_amdgcn_wmma_f32_16x16x32_bf16(false, a, false, b, (short)0, c, false, false);
}

// ---------------------------------------------------------------------------
// fp32 -> bf16 conversion (grid-stride)
// ---------------------------------------------------------------------------
__global__ void f2bf_kernel(const float* __restrict__ in,
                            unsigned short* __restrict__ out, size_t n) {
  size_t i = (size_t)blockIdx.x * blockDim.x + threadIdx.x;
  size_t stride = (size_t)gridDim.x * blockDim.x;
  for (; i < n; i += stride) out[i] = f2bf(in[i]);
}

// ---------------------------------------------------------------------------
// fp32 (K x N) -> bf16 transposed (N x K), 32x32 LDS tile transpose.
// Makes GEMM B tiles identical row copies to A tiles (no LDS scatter).
// ---------------------------------------------------------------------------
__global__ __launch_bounds__(256)
void f2bfT_kernel(const float* __restrict__ in, unsigned short* __restrict__ out,
                  int K, int N) {
  __shared__ unsigned short t[32][33];
  const int kb = blockIdx.y * 32, nb = blockIdx.x * 32;
  const int tx = threadIdx.x & 31, ty = threadIdx.x >> 5;   // ty 0..7
#pragma unroll
  for (int i = 0; i < 32; i += 8)
    t[ty + i][tx] = f2bf(in[(size_t)(kb + ty + i) * N + nb + tx]);
  __syncthreads();
#pragma unroll
  for (int i = 0; i < 32; i += 8)
    out[(size_t)(nb + ty + i) * K + kb + tx] = t[tx][ty + i];
}

// ---------------------------------------------------------------------------
// Tiled bf16 GEMM: C(MxN) = A(MxK) * Bt(NxK)^T, fp32 accumulate in WMMA C.
// Block tile 128x128, K-tile 32, 256 threads = 8 waves (4x2 wave grid).
// Double-buffered LDS, tiles streamed with global_load_async_to_lds_b128.
// out_mode: 0 = f32 row-major, 1 = bf16 row-major,
//           2 = bf16 per-batch transposed (V -> (B, N, Sc) d-major).
// ---------------------------------------------------------------------------
__global__ __launch_bounds__(256)
void gemm_bf16_kernel(const unsigned short* __restrict__ A,
                      const unsigned short* __restrict__ Bt,
                      void* __restrict__ Cout,
                      int M, int N, int K, int out_mode) {
  __shared__ unsigned short sA[2][128][40];   // m-major, 32 K halves + pad
  __shared__ unsigned short sB[2][128][40];   // n-major, 32 K halves + pad

  const int tid  = threadIdx.x;
  const int lane = tid & 31;
  const int wid  = tid >> 5;
  const int wm   = wid & 3;   // 32-row strip
  const int wn   = wid >> 2;  // 64-col strip
  const int m0   = blockIdx.y * 128;
  const int n0   = blockIdx.x * 128;

  v8f acc[2][4];
#pragma unroll
  for (int i = 0; i < 2; ++i)
#pragma unroll
    for (int j = 0; j < 4; ++j)
#pragma unroll
      for (int e = 0; e < 8; ++e) acc[i][j][e] = 0.f;

  auto issue_tile = [&](int buf, int k0) {
#pragma unroll
    for (int i = 0; i < 2; ++i) {
      int e = (tid + i * 256) * 8;
      int r = e >> 5, c = e & 31;
      async_copy_b128(&sA[buf][r][c], A  + (size_t)(m0 + r) * K + k0 + c);
      async_copy_b128(&sB[buf][r][c], Bt + (size_t)(n0 + r) * K + k0 + c);
    }
  };

  const int kTiles = K >> 5;
  issue_tile(0, 0);
  for (int kt = 0; kt < kTiles; ++kt) {
    const int cur = kt & 1;
    wait_async0();        // own async writes to LDS complete
    __syncthreads();      // everyone's complete -> tile `cur` ready
    if (kt + 1 < kTiles) issue_tile(cur ^ 1, (kt + 1) << 5);  // stream next tile

    v16bf af[2];
#pragma unroll
    for (int i = 0; i < 2; ++i)
      af[i] = load_frag(&sA[cur][wm * 32 + i * 16][0], 40, lane);
#pragma unroll
    for (int j = 0; j < 4; ++j) {
      v16bf bfr = load_frag(&sB[cur][wn * 64 + j * 16][0], 40, lane);
#pragma unroll
      for (int i = 0; i < 2; ++i)
        acc[i][j] = wmma_bf16(af[i], bfr, acc[i][j]);
    }
  }

  // C layout: VGPR rr, lanes 0-15 -> M=rr, lanes 16-31 -> M=rr+8; col = lane%16
  const int r  = lane & 15;
  const int hi = lane >> 4;
#pragma unroll
  for (int i = 0; i < 2; ++i)
#pragma unroll
    for (int j = 0; j < 4; ++j)
#pragma unroll
      for (int rr = 0; rr < 8; ++rr) {
        size_t row = (size_t)m0 + wm * 32 + i * 16 + rr + hi * 8;
        size_t col = (size_t)n0 + wn * 64 + j * 16 + r;
        float v = acc[i][j][rr];
        if (out_mode == 0) {
          ((float*)Cout)[row * (size_t)N + col] = v;
        } else if (out_mode == 1) {
          ((unsigned short*)Cout)[row * (size_t)N + col] = f2bf(v);
        } else {  // per-batch transposed: Vt[b][col][s], M == Bc*Sc
          size_t bb = row >> 11;           // row / Sc
          size_t sr = row & (size_t)2047;  // row % Sc
          ((unsigned short*)Cout)[(bb * (size_t)N + col) * Sc + sr] = f2bf(v);
        }
      }
}

// ---------------------------------------------------------------------------
// RoPE: src (B*S, nheads*HD) bf16 -> dst (B, nheads, S, HD) bf16
// ---------------------------------------------------------------------------
__global__ void rope_kernel(const unsigned short* __restrict__ src,
                            const long long* __restrict__ pos,
                            unsigned short* __restrict__ dst,
                            int nheads) {
  size_t idx   = (size_t)blockIdx.x * blockDim.x + threadIdx.x;
  size_t total = (size_t)Bc * Sc * nheads * (HDc / 2);
  if (idx >= total) return;
  int d = (int)(idx & 127);
  size_t t = idx >> 7;
  int hh = (int)(t % nheads); t /= nheads;
  int s  = (int)(t % Sc);
  int b  = (int)(t / Sc);
  size_t row  = (size_t)b * Sc + s;
  int    cols = nheads * HDc;
  float x1 = bf2f(src[row * cols + hh * HDc + d]);
  float x2 = bf2f(src[row * cols + hh * HDc + d + 128]);
  float p  = (float)pos[row];
  // inv_freq = 10000^(-d/128) = exp(-d * ln(1e4)/128)
  float fr = p * __expf(-0.0719558037f * (float)d);
  float c  = __cosf(fr), sn = __sinf(fr);
  size_t ob = (((size_t)b * nheads + hh) * Sc + s) * HDc;
  dst[ob + d]       = f2bf(x1 * c - x2 * sn);
  dst[ob + d + 128] = f2bf(x2 * c + x1 * sn);
}

// ---------------------------------------------------------------------------
// Flash attention (non-causal, additive mask), MQA: one KV head.
// Block = (b, h, 64-query tile); 128 threads = 4 waves, 16 queries per wave.
// K tile staged key-major, V tile comes pre-transposed (d-major) -> both are
// straight async b128 copies. Scores & PV via bf16 WMMA, softmax in fp32.
// ---------------------------------------------------------------------------
__global__ __launch_bounds__(128)
void flash_attn_kernel(const unsigned short* __restrict__ Q,   // (B,NH,S,HD)
                       const unsigned short* __restrict__ Kx,  // (B,S,HD)
                       const unsigned short* __restrict__ Vt,  // (B,HD,S)  d-major
                       const float* __restrict__ mask,         // (B,1,S,S)
                       unsigned short* __restrict__ ctx) {     // (B*S, NH*HD)
  __shared__ unsigned short sK[32][264];    // key-major: B-frags for Q*K^T
  __shared__ unsigned short sV[256][40];    // d-major:   B-frags for P*V
  __shared__ unsigned short sP[4][16][40];  // per-wave P staging (C -> A layout)

  const int tid = threadIdx.x, lane = tid & 31, wid = tid >> 5;
  const int qTiles = Sc / 64;
  const int qt = blockIdx.x % qTiles;
  const int bh = blockIdx.x / qTiles;
  const int h  = bh % NHc, b = bh / NHc;
  const int q0 = qt * 64 + wid * 16;

  const unsigned short* Qbase = Q + (((size_t)b * NHc + h) * Sc + q0) * HDc;
  const unsigned short* Kb  = Kx + (size_t)b * Sc * HDc;
  const unsigned short* Vtb = Vt + (size_t)b * HDc * Sc;
  const float*          mb  = mask + (size_t)b * Sc * Sc;

  // Q fragments (8 x K32) loaded once, directly from global in A-layout.
  v16bf qf[8];
  {
    const int r = lane & 15, hi = lane >> 4;
    const unsigned short* rp = Qbase + (size_t)r * HDc;
#pragma unroll
    for (int kd = 0; kd < 8; ++kd) {
      FragU f;
#pragma unroll
      for (int vv = 0; vv < 8; ++vv) {
        int kofs = kd * 32 + 2 * vv + ((vv >= 4) ? 8 : 0) + (hi ? 8 : 0);
        f.u[vv] = *(const unsigned*)(rp + kofs);
      }
      qf[kd] = f.v;
    }
  }

  v8f acc[16];
#pragma unroll
  for (int nt = 0; nt < 16; ++nt)
#pragma unroll
    for (int e = 0; e < 8; ++e) acc[nt][e] = 0.f;
  float m_i[8], l_i[8];
#pragma unroll
  for (int rr = 0; rr < 8; ++rr) { m_i[rr] = -3.402823466e38f; l_i[rr] = 0.f; }
  const float scale = 0.0625f;      // 1/sqrt(256)

  for (int kt = 0; kt < Sc / 32; ++kt) {
    const int key0 = kt * 32;
    __syncthreads();                // all waves done reading previous tile
    // Async-stream K tile (32x256 key-major) and V tile (256x32 d-major).
#pragma unroll
    for (int i = 0; i < 8; ++i) {
      int e = (tid + i * 128) * 8;
      {
        int kk = e >> 8, dd = e & 255;
        async_copy_b128(&sK[kk][dd], Kb + (size_t)(key0 + kk) * HDc + dd);
      }
      {
        int r = e >> 5, c = e & 31;
        async_copy_b128(&sV[r][c], Vtb + (size_t)r * Sc + key0 + c);
      }
    }
    wait_async0();
    __syncthreads();

    // Scores: 16 queries x 32 keys = two 16x16 C tiles, K-dim = 256 (8 steps)
    v8f s0, s1;
#pragma unroll
    for (int e = 0; e < 8; ++e) { s0[e] = 0.f; s1[e] = 0.f; }
#pragma unroll
    for (int kd = 0; kd < 8; ++kd) {
      v16bf bk0 = load_frag(&sK[0][kd * 32], 264, lane);
      v16bf bk1 = load_frag(&sK[16][kd * 32], 264, lane);
      s0 = wmma_bf16(qf[kd], bk0, s0);
      s1 = wmma_bf16(qf[kd], bk1, s1);
    }

    // Online softmax (rows: VGPR idx + lane half; cols: 16 lanes -> width-16
    // shfl_xor reductions on the wave32 halves).
    {
      const int r = lane & 15, hi = lane >> 4;
      float al[8];
#pragma unroll
      for (int rr = 0; rr < 8; ++rr) {
        int row = q0 + rr + hi * 8;
        float a0 = s0[rr] * scale + mb[(size_t)row * Sc + key0 + r];
        float a1 = s1[rr] * scale + mb[(size_t)row * Sc + key0 + 16 + r];
        float mx = fmaxf(a0, a1);
#pragma unroll
        for (int off = 8; off > 0; off >>= 1) mx = fmaxf(mx, __shfl_xor(mx, off, 16));
        float mnew = fmaxf(m_i[rr], mx);
        al[rr] = __expf(m_i[rr] - mnew);
        m_i[rr] = mnew;
        float p0 = __expf(a0 - mnew);
        float p1 = __expf(a1 - mnew);
        float rs = p0 + p1;
#pragma unroll
        for (int off = 8; off > 0; off >>= 1) rs += __shfl_xor(rs, off, 16);
        l_i[rr] = l_i[rr] * al[rr] + rs;
        // C-layout -> A-layout staging of P (bf16) in wave-private LDS
        sP[wid][rr + hi * 8][r]      = f2bf(p0);
        sP[wid][rr + hi * 8][16 + r] = f2bf(p1);
      }
#pragma unroll
      for (int nt = 0; nt < 16; ++nt)
#pragma unroll
        for (int rr = 0; rr < 8; ++rr) acc[nt][rr] *= al[rr];
    }
    __syncthreads();  // P visible; keeps waves in lockstep for tile reuse

    // ctx(16x256) += P(16x32) @ V(32x256): one WMMA per 16-wide d tile
    v16bf pf = load_frag(&sP[wid][0][0], 40, lane);
#pragma unroll
    for (int nt = 0; nt < 16; ++nt) {
      v16bf vf = load_frag(&sV[nt * 16][0], 40, lane);
      acc[nt] = wmma_bf16(pf, vf, acc[nt]);
    }
  }

  // Normalize and store ctx in (B*S, NH*HD) bf16 layout for the Wo GEMM.
  const int r = lane & 15, hi = lane >> 4;
  float inv[8];
#pragma unroll
  for (int rr = 0; rr < 8; ++rr) inv[rr] = 1.f / l_i[rr];
#pragma unroll
  for (int nt = 0; nt < 16; ++nt)
#pragma unroll
    for (int rr = 0; rr < 8; ++rr) {
      size_t row = (size_t)b * Sc + q0 + rr + hi * 8;
      size_t col = (size_t)h * HDc + nt * 16 + r;
      ctx[row * (size_t)(NHc * HDc) + col] = f2bf(acc[nt][rr] * inv[rr]);
    }
}

// ---------------------------------------------------------------------------
// Host-side orchestration
// ---------------------------------------------------------------------------
extern "C" void kernel_launch(void* const* d_in, const int* in_sizes, int n_in,
                              void* d_out, int out_size, void* d_ws, size_t ws_size,
                              hipStream_t stream) {
  (void)in_sizes; (void)n_in; (void)out_size; (void)ws_size;
  const float*     X    = (const float*)d_in[0];
  const long long* pos  = (const long long*)d_in[1];   // position_ids (int64)
  const float*     mask = (const float*)d_in[2];
  const float*     Wq   = (const float*)d_in[3];
  const float*     Wk   = (const float*)d_in[4];
  const float*     Wv   = (const float*)d_in[5];
  const float*     Wo   = (const float*)d_in[6];

  char* ws = (char*)d_ws;
  size_t off = 0;
  auto alloc = [&](size_t bytes) -> void* {
    void* p = ws + off;
    off += (bytes + 255) & ~(size_t)255;
    return p;
  };
  const size_t MS = (size_t)Bc * Sc;  // 8192 token rows
  unsigned short* Xb   = (unsigned short*)alloc(MS * HIDc * 2);
  unsigned short* WqT  = (unsigned short*)alloc((size_t)NHc * HDc * HIDc * 2); // (2048,2048)
  unsigned short* WkT  = (unsigned short*)alloc((size_t)HDc * HIDc * 2);       // (256,2048)
  unsigned short* WvT  = (unsigned short*)alloc((size_t)HDc * HIDc * 2);
  unsigned short* WoT  = (unsigned short*)alloc((size_t)HIDc * NHc * HDc * 2);
  unsigned short* Qraw = (unsigned short*)alloc(MS * NHc * HDc * 2);
  unsigned short* Kraw = (unsigned short*)alloc(MS * HDc * 2);
  unsigned short* Vt   = (unsigned short*)alloc(MS * HDc * 2);                 // (B,HD,S)
  unsigned short* Qr   = (unsigned short*)alloc(MS * NHc * HDc * 2);
  unsigned short* Kr   = (unsigned short*)alloc(MS * HDc * 2);
  unsigned short* Ctxb = (unsigned short*)alloc(MS * NHc * HDc * 2);

  // 1) fp32 -> bf16: activations straight, weights transposed (N x K)
  f2bf_kernel<<<2048, 256, 0, stream>>>(X, Xb, MS * HIDc);
  f2bfT_kernel<<<dim3((NHc * HDc) / 32, HIDc / 32), 256, 0, stream>>>(Wq, WqT, HIDc, NHc * HDc);
  f2bfT_kernel<<<dim3(HDc / 32, HIDc / 32), 256, 0, stream>>>(Wk, WkT, HIDc, HDc);
  f2bfT_kernel<<<dim3(HDc / 32, HIDc / 32), 256, 0, stream>>>(Wv, WvT, HIDc, HDc);
  f2bfT_kernel<<<dim3(HIDc / 32, (NHc * HDc) / 32), 256, 0, stream>>>(Wo, WoT, NHc * HDc, HIDc);

  // 2) QKV projections (bf16 WMMA, fp32 accum); V written pre-transposed
  gemm_bf16_kernel<<<dim3((NHc * HDc) / 128, MS / 128), 256, 0, stream>>>(
      Xb, WqT, Qraw, (int)MS, NHc * HDc, HIDc, 1);
  gemm_bf16_kernel<<<dim3(HDc / 128, MS / 128), 256, 0, stream>>>(
      Xb, WkT, Kraw, (int)MS, HDc, HIDc, 1);
  gemm_bf16_kernel<<<dim3(HDc / 128, MS / 128), 256, 0, stream>>>(
      Xb, WvT, Vt, (int)MS, HDc, HIDc, 2);

  // 3) RoPE on Q and K
  {
    size_t tq = MS * NHc * (HDc / 2);
    rope_kernel<<<(unsigned)((tq + 255) / 256), 256, 0, stream>>>(Qraw, pos, Qr, NHc);
    size_t tk = MS * NKVc * (HDc / 2);
    rope_kernel<<<(unsigned)((tk + 255) / 256), 256, 0, stream>>>(Kraw, pos, Kr, NKVc);
  }

  // 4) Flash attention (MQA): 1 block per (b, h, 64-query tile)
  flash_attn_kernel<<<Bc * NHc * (Sc / 64), 128, 0, stream>>>(Qr, Kr, Vt, mask, Ctxb);

  // 5) Output projection -> fp32 d_out
  gemm_bf16_kernel<<<dim3(HIDc / 128, MS / 128), 256, 0, stream>>>(
      Ctxb, WoT, d_out, (int)MS, HIDc, HIDc, 0);
}
